// EfficientAttention_1460288880764
// MI455X (gfx1250) — matrix operand order repaired and checked by reading, not compiled, not run on hardware
//
#include <hip/hip_runtime.h>

typedef __attribute__((ext_vector_type(16))) __bf16 v16bf;
typedef __attribute__((ext_vector_type(8)))  float  v8f;
typedef __attribute__((ext_vector_type(4)))  unsigned int uint4v;
typedef __attribute__((ext_vector_type(8)))  int int8v;
typedef __attribute__((ext_vector_type(4)))  int int4v;

#define BB    8
#define CC    256
#define NNPX  16384      // H*W
#define GG    32
#define CPG   (CC/GG)    // 8
#define QKVC  (3*CC)     // 768
#define GEPS  1e-5f

// ---------------------------------------------------------------------------
// Tensor Data Mover: async 2D bf16 tile load (global -> LDS), TENSORcnt-tracked.
// D# group0: [1:0]=count=1 | [63:32]=lds_addr | [120:64]=global_addr | type=2.
// D# group1: data_size=2B, tensor_dim0/1, tile_dim0/1, tensor_dim0_stride.
// Groups 2/3 zero (2D tensor). Issued per-wave (EXEC ignored) by wave 0 only.
// Toolchain uses 6-arg builtin: (g0, g1, g2, g3, g4, cpol).
// ---------------------------------------------------------------------------
__device__ __forceinline__ void tdm_load_2d_bf16(const void* gsrc, void* lds_dst,
                                                 unsigned tile_x, unsigned tile_y,
                                                 unsigned row_stride_elems,
                                                 unsigned tensor_x, unsigned tensor_y) {
    unsigned long long ga = (unsigned long long)(uintptr_t)gsrc;
    unsigned lds = (unsigned)(uintptr_t)lds_dst;       // low 32 bits = LDS byte offset
    uint4v g0;
    g0[0] = 1u;                                        // count=1 (user mode)
    g0[1] = lds;                                       // lds_addr
    g0[2] = (unsigned)(ga & 0xFFFFFFFFu);              // global_addr[31:0]
    g0[3] = (unsigned)((ga >> 32) & 0x1FFFFFFu) | (2u << 30);  // addr[56:32] | type=2
    int8v g1;
    g1[0] = (int)(1u << 16);                           // data_size = 1 -> 2 bytes
    g1[1] = (int)((tensor_x & 0xFFFFu) << 16);         // [63:48] tensor_dim0 lo16
    g1[2] = (int)((tensor_x >> 16) | ((tensor_y & 0xFFFFu) << 16)); // dim0 hi | dim1 lo
    g1[3] = (int)((tensor_y >> 16) | (tile_x << 16));  // dim1 hi | [127:112] tile_dim0
    g1[4] = (int)tile_y;                               // [143:128] tile_dim1 (tile_dim2=0)
    g1[5] = (int)row_stride_elems;                     // tensor_dim0_stride[31:0]
    g1[6] = 0;
    g1[7] = 0;
    int4v z4 = {0, 0, 0, 0};
    int8v z8 = {0, 0, 0, 0, 0, 0, 0, 0};
    __builtin_amdgcn_tensor_load_to_lds(g0, g1, z4, z4, z8, 0);
}

// ---------------------------------------------------------------------------
// WMMA fragment loaders (bf16 16x16x32), per CDNA5 ISA VGPR layouts.
// A tile staged in LDS row-major [M][K] (K contiguous).
// B tile staged in LDS TRANSPOSED [N][K] (K contiguous) so loads are linear.
// ---------------------------------------------------------------------------
__device__ __forceinline__ v16bf frag_a_lds(const __bf16* sm, int m0, int ld) {
    int lane = threadIdx.x & 31;
    const __bf16* row = sm + (size_t)(m0 + (lane & 15)) * ld;
    int kb = (lane >> 4) << 3;           // lanes 16-31 start at K=8
    v16bf a;
#pragma unroll
    for (int i = 0; i < 8; ++i) a[i]     = row[kb + i];        // K 0..7 / 8..15
#pragma unroll
    for (int i = 0; i < 8; ++i) a[8 + i] = row[kb + 16 + i];   // K 16..23 / 24..31
    return a;
}

__device__ __forceinline__ v16bf frag_b_lds(const __bf16* smT, int n0, int ld) {
    int lane = threadIdx.x & 31;
    const __bf16* col = smT + (size_t)(n0 + (lane & 15)) * ld;
    int kb = (lane >> 4) << 3;           // lanes 16-31 start at K=8
    v16bf b;
#pragma unroll
    for (int i = 0; i < 8; ++i) b[i]     = col[kb + i];        // K 0..7 / 8..15
#pragma unroll
    for (int i = 0; i < 8; ++i) b[8 + i] = col[kb + 16 + i];   // K 16..23 / 24..31
    return b;
}

// ---------------------------------------------------------------------------
// 0) One-shot f32 -> bf16 weight conversion (avoids per-block re-conversion).
// ---------------------------------------------------------------------------
__global__ __launch_bounds__(256)
void wconv_kernel(const float* __restrict__ w_qkv, const float* __restrict__ w_proj,
                  __bf16* __restrict__ wq, __bf16* __restrict__ wp) {
    int i = blockIdx.x * 256 + threadIdx.x;
    if (i < QKVC * CC) wq[i] = (__bf16)w_qkv[i];
    if (i < CC * CC)   wp[i] = (__bf16)w_proj[i];
}

// ---------------------------------------------------------------------------
// 1) GroupNorm statistics: one block per (b,g); group chunk is contiguous.
// ---------------------------------------------------------------------------
__global__ __launch_bounds__(256)
void gn_stats_kernel(const float* __restrict__ x, float* __restrict__ stats) {
    int bg = blockIdx.x;                                 // b*GG + g
    const float* p = x + (size_t)bg * CPG * NNPX;
    const int len = CPG * NNPX;                          // 131072
    float s = 0.f, ss = 0.f;
    for (int i = threadIdx.x; i < len; i += 256) {
        float v = p[i]; s += v; ss += v * v;
    }
    __shared__ float sh_s[256], sh_q[256];
    sh_s[threadIdx.x] = s; sh_q[threadIdx.x] = ss;
    __syncthreads();
    for (int off = 128; off > 0; off >>= 1) {
        if (threadIdx.x < off) {
            sh_s[threadIdx.x] += sh_s[threadIdx.x + off];
            sh_q[threadIdx.x] += sh_q[threadIdx.x + off];
        }
        __syncthreads();
    }
    if (threadIdx.x == 0) {
        float mu  = sh_s[0] / (float)len;
        float var = sh_q[0] / (float)len - mu * mu;
        stats[2 * bg]     = mu;
        stats[2 * bg + 1] = rsqrtf(var + GEPS);
    }
}

// ---------------------------------------------------------------------------
// 2) Fused GroupNorm-apply + qkv GEMM:  qkv = w_qkv @ xn + b_qkv  (bf16 out)
//    A tile (bf16 weights) via TDM, overlapped with VALU B-tile normalize.
// ---------------------------------------------------------------------------
__global__ __launch_bounds__(256)
void qkv_gemm_kernel(const float* __restrict__ x, const float* __restrict__ gamma,
                     const float* __restrict__ beta, const __bf16* __restrict__ wq,
                     const float* __restrict__ b_qkv, const float* __restrict__ stats,
                     __bf16* __restrict__ qkv) {
    __shared__ __bf16 sA[128 * 32];      // weight tile [m][k]
    __shared__ __bf16 sB[64 * 32];       // xn tile transposed [n][k]
    const int n0 = blockIdx.x * 64;
    const int m0 = blockIdx.y * 128;
    const int b  = blockIdx.z;
    const int tid = threadIdx.x, wave = tid >> 5;
    v8f acc[4] = {};
    for (int k0 = 0; k0 < CC; k0 += 32) {
        if (wave == 0)
            tdm_load_2d_bf16(wq + (size_t)m0 * CC + k0, sA, 32, 128, CC, CC, QKVC);
        for (int idx = tid; idx < 64 * 32; idx += 256) {
            int j = idx >> 6;            // k-local
            int i = idx & 63;            // n-local
            int c = k0 + j;
            float mu = stats[2 * (b * GG + c / CPG)];
            float rs = stats[2 * (b * GG + c / CPG) + 1];
            float v  = x[((size_t)(b * CC + c)) * NNPX + n0 + i];
            sB[i * 32 + j] = (__bf16)((v - mu) * rs * gamma[c] + beta[c]);
        }
        if (wave == 0) __builtin_amdgcn_s_wait_tensorcnt(0);
        __syncthreads();
        v16bf a = frag_a_lds(sA, wave * 16, 32);
#pragma unroll
        for (int t = 0; t < 4; ++t) {
            v16bf bf = frag_b_lds(sB, t * 16, 32);
            acc[t] = __builtin_amdgcn_wmma_f32_16x16x32_bf16(
                false, a, false, bf, (short)0, acc[t], false, false);
        }
        __syncthreads();
    }
    const int lane = tid & 31;
    const int mb   = m0 + wave * 16 + ((lane >> 4) << 3);
    const int nc   = lane & 15;
#pragma unroll
    for (int t = 0; t < 4; ++t)
#pragma unroll
        for (int r = 0; r < 8; ++r) {
            int m = mb + r, n = n0 + t * 16 + nc;
            qkv[((size_t)(b * QKVC + m)) * NNPX + n] = (__bf16)(acc[t][r] + b_qkv[m]);
        }
}

// ---------------------------------------------------------------------------
// 3) Row softmax over N=16384 (q rows scaled by C^-0.5), in-place bf16.
// ---------------------------------------------------------------------------
__global__ __launch_bounds__(256)
void softmax_kernel(__bf16* __restrict__ qkv) {
    int row = blockIdx.x;                // 0 .. B*512-1  (q then k rows)
    int b = row >> 9, r = row & 511;
    float scale = (r < CC) ? 0.0625f : 1.0f;   // 1/sqrt(256)
    __bf16* p = qkv + ((size_t)(b * QKVC + r)) * NNPX;
    __shared__ float sh[256];
    float mx = -3.0e38f;
    for (int i = threadIdx.x; i < NNPX; i += 256) mx = fmaxf(mx, (float)p[i]);
    sh[threadIdx.x] = mx; __syncthreads();
    for (int off = 128; off > 0; off >>= 1) {
        if (threadIdx.x < off) sh[threadIdx.x] = fmaxf(sh[threadIdx.x], sh[threadIdx.x + off]);
        __syncthreads();
    }
    mx = sh[0]; __syncthreads();
    float s = 0.f;
    for (int i = threadIdx.x; i < NNPX; i += 256) s += __expf((float)p[i] - mx);
    sh[threadIdx.x] = s; __syncthreads();
    for (int off = 128; off > 0; off >>= 1) {
        if (threadIdx.x < off) sh[threadIdx.x] += sh[threadIdx.x + off];
        __syncthreads();
    }
    float inv = scale / sh[0];
    for (int i = threadIdx.x; i < NNPX; i += 256)
        p[i] = (__bf16)(__expf((float)p[i] - mx) * inv);
}

// ---------------------------------------------------------------------------
// 4) context = k @ v^T  (C x C per batch), split-K=8 with f32 atomics.
//    Fully TDM-fed, double-buffered: DMA of tile i+1 overlaps WMMA of tile i.
// ---------------------------------------------------------------------------
__global__ void zero_f32_kernel(float* __restrict__ p, int n) {
    int i = blockIdx.x * blockDim.x + threadIdx.x;
    if (i < n) p[i] = 0.f;
}

__global__ __launch_bounds__(256)
void context_kernel(const __bf16* __restrict__ qkv, float* __restrict__ ctx) {
    __shared__ __bf16 sA[2][64 * 32];    // k tile [c][n]
    __shared__ __bf16 sB[2][64 * 32];    // v tile [d][n]
    const int tile = blockIdx.x;
    const int c0 = (tile >> 2) * 64, d0 = (tile & 3) * 64;
    const int kbeg = blockIdx.y * (NNPX / 8);
    const int b = blockIdx.z;
    const __bf16* kmat = qkv + ((size_t)(b * QKVC + CC))     * NNPX;
    const __bf16* vmat = qkv + ((size_t)(b * QKVC + 2 * CC)) * NNPX;
    const int tid = threadIdx.x, wave = tid >> 5;
    const int msub = wave & 3, npair = wave >> 2;
    const int KT = (NNPX / 8) / 32;      // 64 k-tiles of depth 32
    v8f acc[2] = {};
    if (wave == 0) {
        tdm_load_2d_bf16(kmat + (size_t)c0 * NNPX + kbeg, sA[0], 32, 64, NNPX, NNPX, CC);
        tdm_load_2d_bf16(vmat + (size_t)d0 * NNPX + kbeg, sB[0], 32, 64, NNPX, NNPX, CC);
    }
    for (int it = 0; it < KT; ++it) {
        int cur = it & 1;
        if (wave == 0) __builtin_amdgcn_s_wait_tensorcnt(0);
        __syncthreads();                 // tile `it` resident in buf `cur`
        if (wave == 0 && it + 1 < KT) {  // prefetch tile it+1 into other buffer
            int koff = kbeg + (it + 1) * 32;
            tdm_load_2d_bf16(kmat + (size_t)c0 * NNPX + koff, sA[cur ^ 1], 32, 64, NNPX, NNPX, CC);
            tdm_load_2d_bf16(vmat + (size_t)d0 * NNPX + koff, sB[cur ^ 1], 32, 64, NNPX, NNPX, CC);
        }
        v16bf a = frag_a_lds(sA[cur], msub * 16, 32);
#pragma unroll
        for (int t = 0; t < 2; ++t) {
            v16bf bf = frag_b_lds(sB[cur], (2 * npair + t) * 16, 32);
            acc[t] = __builtin_amdgcn_wmma_f32_16x16x32_bf16(
                false, a, false, bf, (short)0, acc[t], false, false);
        }
        __syncthreads();                 // all reads of buf `cur` done
    }
    const int lane = tid & 31;
    const int mb = c0 + msub * 16 + ((lane >> 4) << 3);
    const int nc = lane & 15;
    float* cb = ctx + (size_t)b * CC * CC;
#pragma unroll
    for (int t = 0; t < 2; ++t) {
        int d = d0 + (2 * npair + t) * 16 + nc;
#pragma unroll
        for (int r = 0; r < 8; ++r)
            atomicAdd(&cb[(mb + r) * CC + d], acc[t][r]);
    }
}

// ---------------------------------------------------------------------------
// 5) Mb = w_proj @ context (tiny 256^3 GEMM per batch), bf16 out.
// ---------------------------------------------------------------------------
__global__ __launch_bounds__(256)
void projctx_kernel(const __bf16* __restrict__ wp, const float* __restrict__ ctx,
                    __bf16* __restrict__ Mb) {
    __shared__ __bf16 sA[64 * 32];       // w_proj [o][c]
    __shared__ __bf16 sB[64 * 32];       // ctx^T  [d][c]
    const int tile = blockIdx.x;
    const int o0 = (tile >> 2) * 64, d0 = (tile & 3) * 64;
    const int b = blockIdx.y;
    const float* cb = ctx + (size_t)b * CC * CC;
    const int tid = threadIdx.x, wave = tid >> 5;
    const int msub = wave & 3, npair = wave >> 2;
    v8f acc[2] = {};
    for (int k0 = 0; k0 < CC; k0 += 32) {
        for (int idx = tid; idx < 64 * 32; idx += 256) {
            int i = idx >> 5, j = idx & 31;
            sA[idx] = wp[(o0 + i) * CC + k0 + j];
            sB[idx] = (__bf16)cb[(k0 + j) * CC + d0 + i];   // transpose in staging
        }
        __syncthreads();
        v16bf a = frag_a_lds(sA, msub * 16, 32);
#pragma unroll
        for (int t = 0; t < 2; ++t) {
            v16bf bf = frag_b_lds(sB, (2 * npair + t) * 16, 32);
            acc[t] = __builtin_amdgcn_wmma_f32_16x16x32_bf16(
                false, a, false, bf, (short)0, acc[t], false, false);
        }
        __syncthreads();
    }
    const int lane = tid & 31;
    const int mb = o0 + msub * 16 + ((lane >> 4) << 3);
    const int nc = lane & 15;
    __bf16* out = Mb + (size_t)b * CC * CC;
#pragma unroll
    for (int t = 0; t < 2; ++t) {
        int d = d0 + (2 * npair + t) * 16 + nc;
#pragma unroll
        for (int r = 0; r < 8; ++r)
            out[(mb + r) * CC + d] = (__bf16)acc[t][r];
    }
}

// ---------------------------------------------------------------------------
// 6) out = Mb @ q + b_proj + x   (f32 output). A tile via TDM.
// ---------------------------------------------------------------------------
__global__ __launch_bounds__(256)
void out_gemm_kernel(const __bf16* __restrict__ Mb, const __bf16* __restrict__ qkv,
                     const float* __restrict__ b_proj, const float* __restrict__ x,
                     float* __restrict__ out) {
    __shared__ __bf16 sA[128 * 32];      // Mb [o][c]
    __shared__ __bf16 sB[64 * 32];       // q^T [n][c]
    const int n0 = blockIdx.x * 64;
    const int m0 = blockIdx.y * 128;
    const int b  = blockIdx.z;
    const __bf16* A = Mb + (size_t)b * CC * CC;
    const __bf16* q = qkv + ((size_t)(b * QKVC)) * NNPX;     // q block at offset 0
    const int tid = threadIdx.x, wave = tid >> 5;
    v8f acc[4] = {};
    for (int k0 = 0; k0 < CC; k0 += 32) {
        if (wave == 0)
            tdm_load_2d_bf16(A + (size_t)m0 * CC + k0, sA, 32, 128, CC, CC, CC);
        for (int idx = tid; idx < 64 * 32; idx += 256) {
            int j = idx >> 6, i = idx & 63;
            sB[i * 32 + j] = q[((size_t)(k0 + j)) * NNPX + n0 + i];
        }
        if (wave == 0) __builtin_amdgcn_s_wait_tensorcnt(0);
        __syncthreads();
        v16bf a = frag_a_lds(sA, wave * 16, 32);
#pragma unroll
        for (int t = 0; t < 4; ++t) {
            v16bf bf = frag_b_lds(sB, t * 16, 32);
            acc[t] = __builtin_amdgcn_wmma_f32_16x16x32_bf16(
                false, a, false, bf, (short)0, acc[t], false, false);
        }
        __syncthreads();
    }
    const int lane = tid & 31;
    const int mb = m0 + wave * 16 + ((lane >> 4) << 3);
    const int nc = lane & 15;
#pragma unroll
    for (int t = 0; t < 4; ++t)
#pragma unroll
        for (int r = 0; r < 8; ++r) {
            int m = mb + r, n = n0 + t * 16 + nc;
            size_t off = ((size_t)(b * CC + m)) * NNPX + n;
            out[off] = acc[t][r] + b_proj[m] + x[off];
        }
}

// ---------------------------------------------------------------------------
extern "C" void kernel_launch(void* const* d_in, const int* in_sizes, int n_in,
                              void* d_out, int out_size, void* d_ws, size_t ws_size,
                              hipStream_t stream) {
    const float* x      = (const float*)d_in[0];
    const float* gamma  = (const float*)d_in[1];
    const float* beta   = (const float*)d_in[2];
    const float* w_qkv  = (const float*)d_in[3];
    const float* b_qkv  = (const float*)d_in[4];
    const float* w_proj = (const float*)d_in[5];
    const float* b_proj = (const float*)d_in[6];
    float* out = (float*)d_out;

    // workspace layout
    char* ws = (char*)d_ws;
    size_t off = 0;
    auto up = [](size_t v) { return (v + 255) & ~(size_t)255; };
    __bf16* qkv = (__bf16*)(ws + off); off += up((size_t)BB * QKVC * NNPX * sizeof(__bf16));
    float*  ctx = (float*) (ws + off); off += up((size_t)BB * CC * CC * sizeof(float));
    __bf16* Mb  = (__bf16*)(ws + off); off += up((size_t)BB * CC * CC * sizeof(__bf16));
    float*  st  = (float*) (ws + off); off += up((size_t)BB * GG * 2 * sizeof(float));
    __bf16* wq  = (__bf16*)(ws + off); off += up((size_t)QKVC * CC * sizeof(__bf16));
    __bf16* wp  = (__bf16*)(ws + off); off += up((size_t)CC * CC * sizeof(__bf16));
    (void)ws_size; (void)n_in; (void)in_sizes; (void)out_size;

    wconv_kernel<<<(QKVC * CC + 255) / 256, 256, 0, stream>>>(w_qkv, w_proj, wq, wp);
    gn_stats_kernel<<<BB * GG, 256, 0, stream>>>(x, st);
    qkv_gemm_kernel<<<dim3(NNPX / 64, QKVC / 128, BB), 256, 0, stream>>>(
        x, gamma, beta, wq, b_qkv, st, qkv);
    softmax_kernel<<<BB * 512, 256, 0, stream>>>(qkv);
    int nctx = BB * CC * CC;
    zero_f32_kernel<<<(nctx + 255) / 256, 256, 0, stream>>>(ctx, nctx);
    context_kernel<<<dim3(16, 8, BB), 256, 0, stream>>>(qkv, ctx);
    projctx_kernel<<<dim3(16, BB), 256, 0, stream>>>(wp, ctx, Mb);
    out_gemm_kernel<<<dim3(NNPX / 64, CC / 128, BB), 256, 0, stream>>>(
        Mb, qkv, b_proj, x, out);
}
